// L2_23055384445145
// MI455X (gfx1250) — matrix operand order repaired and checked by reading, not compiled, not run on hardware
//
#include <hip/hip_runtime.h>
#include <hip/hip_bf16.h>
#include <stdint.h>

// Problem constants (match reference setup_inputs)
#define MB   1024   // rows of x
#define NB   4096   // rows of y
#define DK   128    // feature dim (GEMM K)
#define TM   128    // M tile per workgroup
#define TN   128    // N tile per workgroup
#define LDSK 144    // padded LDS row stride (elements): 288B rows -> bank stagger

typedef __attribute__((ext_vector_type(16))) __bf16 v16bf;
typedef __attribute__((ext_vector_type(8)))  __bf16 v8bf;
typedef __attribute__((ext_vector_type(8)))  float  v8f;

// ---------- fp32 -> bf16 split (round-to-nearest-even, hi + lo) ----------
__device__ __forceinline__ uint16_t f32_to_bf16(float f) {
    uint32_t u = __float_as_uint(f);
    u += 0x7FFFu + ((u >> 16) & 1u);
    return (uint16_t)(u >> 16);
}
__device__ __forceinline__ float bf16_to_f32(uint16_t h) {
    return __uint_as_float(((uint32_t)h) << 16);
}
__device__ __forceinline__ void split_store4(float4 v, uint16_t* hp, uint16_t* lp) {
    float f[4] = {v.x, v.y, v.z, v.w};
    uint16_t h[4], l[4];
#pragma unroll
    for (int j = 0; j < 4; ++j) {
        h[j] = f32_to_bf16(f[j]);
        l[j] = f32_to_bf16(f[j] - bf16_to_f32(h[j]));
    }
    uint2 hv, lv;
    hv.x = (uint32_t)h[0] | ((uint32_t)h[1] << 16);
    hv.y = (uint32_t)h[2] | ((uint32_t)h[3] << 16);
    lv.x = (uint32_t)l[0] | ((uint32_t)l[1] << 16);
    lv.y = (uint32_t)l[2] | ((uint32_t)l[3] << 16);
    *(uint2*)hp = hv;
    *(uint2*)lp = lv;
}

// ---------- fragment loads from LDS (bf16, 16x16x32 WMMA layouts) ----------
// A (16x32): lanes 0-15 -> M=lane, K {0..7,16..23}; lanes 16-31 -> M=lane-16, K {8..15,24..31}
__device__ __forceinline__ v16bf load_afrag(const uint16_t* tile, int row, int kk, int g) {
    const uint16_t* p = tile + row * LDSK + kk + g * 8;
    union { v16bf v; v8bf h[2]; } u;
    u.h[0] = *(const v8bf*)(p);
    u.h[1] = *(const v8bf*)(p + 16);
    return u.v;
}
// B (32x16): lanes 0-15 -> N=lane, K {0..7,16..23}; lanes 16-31 -> N=lane-16, K {8..15,24..31}
__device__ __forceinline__ v16bf load_bfrag(const uint16_t* tile, int row, int kk, int g) {
    const uint16_t* p = tile + row * LDSK + kk + g * 8;
    union { v16bf v; v8bf h[2]; } u;
    u.h[0] = *(const v8bf*)(p);
    u.h[1] = *(const v8bf*)(p + 16);
    return u.v;
}

// ---------- kernel 1: exact fp32 row norms into workspace ----------
__global__ void __launch_bounds__(256) l2_norms_kernel(const float* __restrict__ x,
                                                       const float* __restrict__ y,
                                                       float* __restrict__ norms) {
    int tid = blockIdx.x * blockDim.x + threadIdx.x;   // 0..5119
    if (tid >= MB + NB) return;
    const float* rowp = (tid < MB) ? (x + (size_t)tid * DK)
                                   : (y + (size_t)(tid - MB) * DK);
    const float4* r4 = (const float4*)rowp;
    float s = 0.f;
#pragma unroll
    for (int i = 0; i < DK / 4; ++i) {
        float4 v = r4[i];
        s += v.x * v.x + v.y * v.y + v.z * v.z + v.w * v.w;
    }
    norms[tid] = s;
}

// ---------- kernel 2: tiled bf16x3 WMMA GEMM with fused distance epilogue ----------
extern __shared__ uint16_t smem[];

__global__ void __launch_bounds__(256) l2_wmma_kernel(const float* __restrict__ X,
                                                      const float* __restrict__ Y,
                                                      const float* __restrict__ norms,
                                                      float* __restrict__ out) {
    uint16_t* Xh = smem;                 // TM x LDSK
    uint16_t* Xl = Xh + TM * LDSK;
    uint16_t* Yh = Xl + TM * LDSK;
    uint16_t* Yl = Yh + TN * LDSK;

    const int m0  = blockIdx.y * TM;
    const int n0  = blockIdx.x * TN;
    const int tid = threadIdx.x;

    // --- stage X tile (128 rows x 128 K, fp32 -> bf16 hi/lo) ---
#pragma unroll
    for (int i = 0; i < (TM * DK / 4) / 256; ++i) {
        int flat = tid + i * 256;          // float4 index within tile
        int row  = flat >> 5;              // 32 float4 per row
        int c4   = flat & 31;
        float4 v = ((const float4*)(X + (size_t)(m0 + row) * DK))[c4];
        split_store4(v, Xh + row * LDSK + c4 * 4, Xl + row * LDSK + c4 * 4);
    }
    // --- stage Y tile ---
#pragma unroll
    for (int i = 0; i < (TN * DK / 4) / 256; ++i) {
        int flat = tid + i * 256;
        int row  = flat >> 5;
        int c4   = flat & 31;
        float4 v = ((const float4*)(Y + (size_t)(n0 + row) * DK))[c4];
        split_store4(v, Yh + row * LDSK + c4 * 4, Yl + row * LDSK + c4 * 4);
    }
    __syncthreads();

    const int lane = tid & 31;
    const int w    = tid >> 5;          // wave id 0..7
    const int wm   = (w & 1) * 64;      // M offset of this wave in tile
    const int wn   = (w >> 1) * 32;     // N offset of this wave in tile
    const int g    = lane >> 4;         // lane half (K-group select)
    const int l16  = lane & 15;

    v8f acc[4][2] = {};                 // 4 M-subtiles x 2 N-subtiles of 16x16 f32

#pragma unroll
    for (int kk = 0; kk < DK; kk += 32) {
        v16bf ah[4], al[4], bh[2], bl[2];
#pragma unroll
        for (int mi = 0; mi < 4; ++mi) {
            int r = wm + mi * 16 + l16;
            ah[mi] = load_afrag(Xh, r, kk, g);
            al[mi] = load_afrag(Xl, r, kk, g);
        }
#pragma unroll
        for (int ni = 0; ni < 2; ++ni) {
            int c = wn + ni * 16 + l16;
            bh[ni] = load_bfrag(Yh, c, kk, g);
            bl[ni] = load_bfrag(Yl, c, kk, g);
        }
#pragma unroll
        for (int mi = 0; mi < 4; ++mi) {
#pragma unroll
            for (int ni = 0; ni < 2; ++ni) {
                // bf16x3 fp32 emulation: hi*hi + hi*lo + lo*hi
                acc[mi][ni] = __builtin_amdgcn_wmma_f32_16x16x32_bf16(
                    false, ah[mi], false, bh[ni], (short)0, acc[mi][ni], false, false);
                acc[mi][ni] = __builtin_amdgcn_wmma_f32_16x16x32_bf16(
                    false, ah[mi], false, bl[ni], (short)0, acc[mi][ni], false, false);
                acc[mi][ni] = __builtin_amdgcn_wmma_f32_16x16x32_bf16(
                    false, al[mi], false, bh[ni], (short)0, acc[mi][ni], false, false);
            }
        }
    }

    // --- epilogue: out = 2*dot - ||x||^2 - ||y||^2 ---
    const float* xn = norms;
    const float* yn = norms + MB;
#pragma unroll
    for (int mi = 0; mi < 4; ++mi) {
        const int rbase = m0 + wm + mi * 16 + g * 8;   // C layout: VGPR v -> M = v + g*8
        float xns[8];
#pragma unroll
        for (int v = 0; v < 8; ++v) xns[v] = xn[rbase + v];
#pragma unroll
        for (int ni = 0; ni < 2; ++ni) {
            const int col = n0 + wn + ni * 16 + l16;   // C layout: N = lane&15
            const float ynv = yn[col];
#pragma unroll
            for (int v = 0; v < 8; ++v) {
                out[(size_t)(rbase + v) * NB + col] = 2.0f * acc[mi][ni][v] - xns[v] - ynv;
            }
        }
    }
}

extern "C" void kernel_launch(void* const* d_in, const int* in_sizes, int n_in,
                              void* d_out, int out_size, void* d_ws, size_t ws_size,
                              hipStream_t stream) {
    (void)in_sizes; (void)n_in; (void)out_size; (void)ws_size;
    const float* x = (const float*)d_in[0];   // [1024,128]
    const float* y = (const float*)d_in[1];   // [4096,128]
    float* norms   = (float*)d_ws;            // 5120 floats: [xnorm | ynorm]
    float* out     = (float*)d_out;           // [1024,4096]

    l2_norms_kernel<<<(MB + NB + 255) / 256, 256, 0, stream>>>(x, y, norms);

    dim3 grid(NB / TN, MB / TM);              // (32, 8)
    size_t smem_bytes = (size_t)(2 * TM + 2 * TN) * LDSK * sizeof(uint16_t); // 147456 B
    l2_wmma_kernel<<<grid, 256, smem_bytes, stream>>>(x, y, norms, out);
}
